// SwinTransformer_231928234308
// MI455X (gfx1250) — matrix operand order repaired and checked by reading, not compiled, not run on hardware
//
#include <hip/hip_runtime.h>
#include <hip/hip_bf16.h>
#include <cstdint>

typedef __attribute__((ext_vector_type(16))) _Float16 v16h;
typedef __attribute__((ext_vector_type(8)))  _Float16 v8h;
typedef __attribute__((ext_vector_type(8)))  float    v8f;
typedef unsigned int v4u __attribute__((ext_vector_type(4)));
typedef int          v8i __attribute__((ext_vector_type(8)));
typedef int          v4i __attribute__((ext_vector_type(4)));

#define C_DIM   96
#define NHEADS  3
#define NTOK    64
#define GRID_W  12
#define QK_SCALE 0.17677669529663687f   // (96/3)^-0.5
#define LN_EPSF  1e-6f

// packed weights in d_ws (f16): per head h, 4 contiguous Bt blocks of 9216 halves:
//   [h*36864 + 0     ] Wq[d*96+kc],  [.. + 9216 ] Wk,  [.. +18432] Wv,  [.. +27648] Wo
#define PACK_HALVES_PER_HEAD 36864
#define PACK_TOTAL_HALVES    (PACK_HALVES_PER_HEAD * NHEADS)   // 110592 -> 221184 bytes

#if defined(__has_builtin)
#  if __has_builtin(__builtin_amdgcn_tensor_load_to_lds) && __has_builtin(__builtin_amdgcn_s_wait_tensorcnt)
#    define USE_TDM 1
#  endif
#endif
#ifndef USE_TDM
#  define USE_TDM 0
#endif

// ---- WMMA operand gathers (CDNA5 16-bit layouts, wave32) ----
// A (16x32 f16): lane L -> M = L&15 ; K chunks [kh*8..+7] and [16+kh*8..+7], kh = L>>4
static __device__ __forceinline__ v16h ldsA(const _Float16* __restrict__ base, int stride,
                                            int mBase, int kBase, int lane) {
    const _Float16* p = base + (mBase + (lane & 15)) * stride + kBase + ((lane >> 4) << 3);
    v8h lo = *(const v8h*)(p);
    v8h hi = *(const v8h*)(p + 16);
    return __builtin_shufflevector(lo, hi, 0,1,2,3,4,5,6,7,8,9,10,11,12,13,14,15);
}
// B (32x16 f16) from Bt[N][K] row-major: lane L -> N = L&15 ; K chunks [kh*8..+7] and [16+kh*8..+7], kh = L>>4
static __device__ __forceinline__ v16h ldsB(const _Float16* __restrict__ base, int stride,
                                            int nBase, int kBase, int lane) {
    const _Float16* p = base + (nBase + (lane & 15)) * stride + kBase + ((lane >> 4) << 3);
    v8h lo = *(const v8h*)(p);
    v8h hi = *(const v8h*)(p + 16);
    return __builtin_shufflevector(lo, hi, 0,1,2,3,4,5,6,7,8,9,10,11,12,13,14,15);
}

static __device__ __forceinline__ v8f wmma16(v16h a, v16h b, v8f c) {
    return __builtin_amdgcn_wmma_f32_16x16x32_f16(false, a, false, b, (short)0, c, false, false);
}

#if USE_TDM
// One TDM descriptor: contiguous copy of n8*8 bytes global -> LDS (1-D tile, 8B elements).
// D# per ISA ch.8: group0 = {count=1, lds_addr, global_addr, type=2}; group1 packs
// data_size=3(8B), tensor_dim0=n8, tensor_dim1=1, tile_dim0=n8, tile_dim1=1, dim0_stride=n8.
static __device__ __forceinline__ void tdm_copy_to_lds(const void* gsrc, unsigned lds_off,
                                                       unsigned n8) {
    unsigned long long ga = (unsigned long long)(uintptr_t)gsrc;
    v4u g0;
    g0[0] = 1u;                                            // count=1, user mode
    g0[1] = lds_off;                                       // lds_addr (bytes)
    g0[2] = (unsigned)(ga & 0xFFFFFFFFull);                // global_addr[31:0]
    g0[3] = (unsigned)((ga >> 32) & 0x01FFFFFFull) | (2u << 30);  // addr[56:32] | type=2
    v8i g1;
    g1[0] = (int)(3u << 16);                               // workgroup_mask=0, data_size=3 (8B)
    g1[1] = (int)((n8 & 0xFFFFu) << 16);                   // tensor_dim0[15:0]
    g1[2] = (int)((n8 >> 16) | (1u << 16));                // tensor_dim0[31:16] | tensor_dim1=1
    g1[3] = (int)((n8 & 0xFFFFu) << 16);                   // tile_dim0
    g1[4] = 1;                                             // tile_dim1=1, tile_dim2=0
    g1[5] = (int)n8;                                       // tensor_dim0_stride[31:0]
    g1[6] = 0;                                             // stride hi, dim1_stride lo
    g1[7] = 0;
    v4i z4 = {0, 0, 0, 0};
    v8i z8 = {0, 0, 0, 0, 0, 0, 0, 0};
    // 6-arg form (clang-23 / therock-10.0 headers): (g0, g1, g2, g3, g?, cpol)
    __builtin_amdgcn_tensor_load_to_lds(g0, g1, z4, z4, z8, 0);
}
#endif

// ---------------- prepack kernel: f32 weights -> f16 Bt-blocks in d_ws ----------------
__global__ __launch_bounds__(256)
void swin_pack_weights(const float* __restrict__ wqkv, const float* __restrict__ wout,
                       _Float16* __restrict__ wpack) {
    int idx = blockIdx.x * 256 + threadIdx.x;
    if (idx >= PACK_TOTAL_HALVES) return;
    const int h  = idx / PACK_HALVES_PER_HEAD;
    const int r  = idx - h * PACK_HALVES_PER_HEAD;
    const int m  = r / 9216;              // 0=q 1=k 2=v 3=out
    const int r2 = r - m * 9216;
    const int d  = r2 / 96;               // output channel (Bt row)
    const int kc = r2 - d * 96;           // input channel  (Bt col)
    float v;
    if (m < 3) v = wqkv[kc * 864 + d * 9 + m * 3 + h];   // qkv col = d*9 + kk*3 + h
    else       v = wout[(h * 96 + kc) * 96 + d];
    wpack[idx] = (_Float16)v;
}

// ---------------- main kernel: one workgroup = one 4x4x4 window, 4 wave32 ----------------
template <bool PACKED>
__global__ __launch_bounds__(128)
void swin_win_attn(const float* __restrict__ x,
                   const float* __restrict__ lnw,  const float* __restrict__ lnb,
                   const float* __restrict__ wqkv, const float* __restrict__ rpbt,
                   const float* __restrict__ wout, const _Float16* __restrict__ wpack,
                   float* __restrict__ out)
{
    extern __shared__ float4 dynsmem[];
    _Float16* s_W    = (_Float16*)dynsmem;        // 36864 halves: Wq|Wk|Wv|Wo (Bt[96][96] each)
    _Float16* s_xn   = s_W   + PACK_HALVES_PER_HEAD;   // 64x96
    _Float16* s_q    = s_xn  + NTOK * C_DIM;           // 64x96 (q, later outh)
    _Float16* s_k    = s_q   + NTOK * C_DIM;           // 64x96
    _Float16* s_vT   = s_k   + NTOK * C_DIM;           // 96x64 (v transposed)
    _Float16* s_attn = s_vT  + C_DIM * NTOK;           // 64x64 (scores -> attn)
    float*    s_b7   = (float*)(s_attn + NTOK * NTOK); // 7 bias values for current head

    const int tid  = threadIdx.x;
    const int lane = tid & 31;
    const int wv   = tid >> 5;

    // window decode (matches window_partition ordering: b, D-blk, H-blk, W-blk)
    const int bw = blockIdx.x;
    const int wb = bw / (GRID_W * GRID_W * GRID_W);
    const int r3 = bw % (GRID_W * GRID_W * GRID_W);
    const int wd = r3 / (GRID_W * GRID_W);
    const int wh = (r3 / GRID_W) % GRID_W;
    const int ww = r3 % GRID_W;

    // ---------------- LN: load token row, layernorm, f16 to LDS ----------------
    {
        const int t  = tid >> 1;            // token, two threads per token
        const int hf = tid & 1;             // channel half
        const int ld = t >> 4, lh = (t >> 2) & 3, lw = t & 3;
        const size_t row = ((size_t)((wb * 48 + wd * 4 + ld) * 48 + (wh * 4 + lh)) * 48
                            + (ww * 4 + lw)) * (size_t)C_DIM;
        const float4* px = (const float4*)(x + row + hf * 48);
        float vals[48];
        float s = 0.f, s2 = 0.f;
        #pragma unroll
        for (int i = 0; i < 12; ++i) {
            float4 v = px[i];
            vals[4*i+0] = v.x; vals[4*i+1] = v.y; vals[4*i+2] = v.z; vals[4*i+3] = v.w;
            s  += v.x + v.y + v.z + v.w;
            s2 += v.x*v.x + v.y*v.y + v.z*v.z + v.w*v.w;
        }
        s  += __shfl_xor(s,  1, 32);
        s2 += __shfl_xor(s2, 1, 32);
        const float mean = s  * (1.0f / 96.0f);
        const float var  = s2 * (1.0f / 96.0f) - mean * mean;
        const float rstd = rsqrtf(var + LN_EPSF);
        #pragma unroll
        for (int i = 0; i < 48; ++i) {
            const int ch = hf * 48 + i;
            s_xn[t * C_DIM + ch] = (_Float16)((vals[i] - mean) * rstd * lnw[ch] + lnb[ch]);
        }
    }
    __syncthreads();

    // persistent out-projection accumulators (6 D-tiles per wave, live across heads)
    v8f acc_out[6];
    #pragma unroll
    for (int j = 0; j < 6; ++j) acc_out[j] = (v8f){0.f,0.f,0.f,0.f,0.f,0.f,0.f,0.f};

    for (int h = 0; h < NHEADS; ++h) {
        // ------- stage all 4 weight blocks for this head into s_W -------
        if constexpr (PACKED) {
#if USE_TDM
            if (wv == 0) {   // one TDM op: 73,728 B contiguous global -> LDS
                tdm_copy_to_lds(wpack + (size_t)h * PACK_HALVES_PER_HEAD,
                                (unsigned)(uintptr_t)s_W, PACK_HALVES_PER_HEAD / 4);
                __builtin_amdgcn_s_wait_tensorcnt(0);
            }
#else
            {
                const uint4* src = (const uint4*)(wpack + (size_t)h * PACK_HALVES_PER_HEAD);
                uint4* dst = (uint4*)s_W;
                for (int i = tid; i < PACK_HALVES_PER_HEAD / 8; i += 128) dst[i] = src[i];
            }
#endif
        } else {
            for (int i = tid; i < 9216; i += 128) {
                const int d = i / 96, kc = i - d * 96;
                s_W[i]         = (_Float16)wqkv[kc * 864 + d * 9 + h];
                s_W[9216 + i]  = (_Float16)wqkv[kc * 864 + d * 9 + 3 + h];
                s_W[18432 + i] = (_Float16)wqkv[kc * 864 + d * 9 + 6 + h];
                s_W[27648 + i] = (_Float16)wout[(h * 96 + kc) * 96 + d];
            }
        }
        if (tid < 7) s_b7[tid] = rpbt[tid * NHEADS + h];
        __syncthreads();

        // ------- G1: q = xn@Wq, k = xn@Wk  (48 tile-jobs, 12 per wave) -------
        for (int j = 0; j < 12; ++j) {
            const int job = wv + 4 * j;                  // wave-uniform
            const int t24 = (job < 24) ? job : (job - 24);
            const _Float16* W   = (job < 24) ? s_W : (s_W + 9216);
            _Float16*       dst = (job < 24) ? s_q : s_k;
            const int mt = t24 / 6, nt = t24 - mt * 6;
            v8f acc = (v8f){0.f,0.f,0.f,0.f,0.f,0.f,0.f,0.f};
            #pragma unroll
            for (int ks = 0; ks < 3; ++ks) {
                v16h a = ldsA(s_xn, C_DIM, mt * 16, ks * 32, lane);
                v16h b = ldsB(W,    C_DIM, nt * 16, ks * 32, lane);
                acc = wmma16(a, b, acc);
            }
            const int col  = nt * 16 + (lane & 15);
            const int row0 = mt * 16 + ((lane >> 4) << 3);
            #pragma unroll
            for (int r = 0; r < 8; ++r)
                dst[(row0 + r) * C_DIM + col] = (_Float16)acc[r];
        }
        __syncthreads();

        // ------- G2: v = xn@Wv stored transposed vT[ch][tok] (contiguous v8h store) -------
        for (int j = 0; j < 6; ++j) {
            const int tt = wv + 4 * j;
            const int mt = tt / 6, nt = tt - mt * 6;
            v8f acc = (v8f){0.f,0.f,0.f,0.f,0.f,0.f,0.f,0.f};
            #pragma unroll
            for (int ks = 0; ks < 3; ++ks) {
                v16h a = ldsA(s_xn,        C_DIM, mt * 16, ks * 32, lane);
                v16h b = ldsB(s_W + 18432, C_DIM, nt * 16, ks * 32, lane);
                acc = wmma16(a, b, acc);
            }
            const int ch   = nt * 16 + (lane & 15);
            const int tok0 = mt * 16 + ((lane >> 4) << 3);
            v8h hv = {};
            #pragma unroll
            for (int r = 0; r < 8; ++r) hv[r] = (_Float16)acc[r];
            *(v8h*)(s_vT + ch * NTOK + tok0) = hv;
        }

        // ------- S: scores = q@k^T * scale + bias -> f16 s_attn (disjoint from G2) -------
        for (int j = 0; j < 4; ++j) {
            const int tt = wv + 4 * j;
            const int it = tt >> 2, jt = tt & 3;
            v8f acc = (v8f){0.f,0.f,0.f,0.f,0.f,0.f,0.f,0.f};
            #pragma unroll
            for (int ks = 0; ks < 3; ++ks) {
                v16h a = ldsA(s_q, C_DIM, it * 16, ks * 32, lane);
                v16h b = ldsB(s_k, C_DIM, jt * 16, ks * 32, lane);  // k rows == Bt for q.k^T
                acc = wmma16(a, b, acc);
            }
            const int col  = jt * 16 + (lane & 15);
            const int row0 = it * 16 + ((lane >> 4) << 3);
            const int gj = (col < 32) ? ((col >> 2) & 3) : (col & 3);
            #pragma unroll
            for (int r = 0; r < 8; ++r) {
                const int i  = row0 + r;
                const int gi = (i < 32) ? ((i >> 2) & 3) : (i & 3);
                int rel = (i >> 4) - (col >> 4) + gi - gj;   // faithful (2,-1)-reshape index
                rel = rel < 0 ? 0 : rel;                     // clip low; max 6 < 342
                s_attn[i * NTOK + col] = (_Float16)(acc[r] * QK_SCALE + s_b7[rel]);
            }
        }
        __syncthreads();

        // ------- SM: row softmax (2 lanes per row, f32 math) -------
        {
            const int row = tid >> 1, hf = tid & 1;
            _Float16* arow = s_attn + row * NTOK + hf * 32;
            float vals[32];
            float mx = -3.0e38f;
            #pragma unroll
            for (int i = 0; i < 32; ++i) { vals[i] = (float)arow[i]; mx = fmaxf(mx, vals[i]); }
            mx = fmaxf(mx, __shfl_xor(mx, 1, 32));
            float sum = 0.f;
            #pragma unroll
            for (int i = 0; i < 32; ++i) { vals[i] = __expf(vals[i] - mx); sum += vals[i]; }
            sum += __shfl_xor(sum, 1, 32);
            const float inv = 1.0f / sum;
            #pragma unroll
            for (int i = 0; i < 32; ++i) arow[i] = (_Float16)(vals[i] * inv);
        }
        __syncthreads();

        // ------- AV: outh = attn @ v (B from vT rows); outh overwrites s_q -------
        for (int j = 0; j < 6; ++j) {
            const int tt = wv + 4 * j;
            const int mt = tt / 6, nt = tt - mt * 6;
            v8f acc = (v8f){0.f,0.f,0.f,0.f,0.f,0.f,0.f,0.f};
            #pragma unroll
            for (int ks = 0; ks < 2; ++ks) {
                v16h a = ldsA(s_attn, NTOK, mt * 16, ks * 32, lane);
                v16h b = ldsB(s_vT,   NTOK, nt * 16, ks * 32, lane);
                acc = wmma16(a, b, acc);
            }
            const int col  = nt * 16 + (lane & 15);
            const int row0 = mt * 16 + ((lane >> 4) << 3);
            #pragma unroll
            for (int r = 0; r < 8; ++r)
                s_q[(row0 + r) * C_DIM + col] = (_Float16)acc[r];
        }
        __syncthreads();

        // ------- P: out += outh @ Wout_h (accumulate in VGPRs across heads) -------
        #pragma unroll
        for (int j = 0; j < 6; ++j) {
            const int tt = wv + 4 * j;
            const int mt = tt / 6, nt = tt - mt * 6;
            #pragma unroll
            for (int ks = 0; ks < 3; ++ks) {
                v16h a = ldsA(s_q,         C_DIM, mt * 16, ks * 32, lane);
                v16h b = ldsB(s_W + 27648, C_DIM, nt * 16, ks * 32, lane);
                acc_out[j] = wmma16(a, b, acc_out[j]);
            }
        }
        __syncthreads();   // protect s_W / s_q before next head rewrites them
    }

    // ---------------- final store with window reverse ----------------
    #pragma unroll
    for (int j = 0; j < 6; ++j) {
        const int tt = wv + 4 * j;
        const int mt = tt / 6, nt = tt - mt * 6;
        const int col  = nt * 16 + (lane & 15);
        const int tok0 = mt * 16 + ((lane >> 4) << 3);
        #pragma unroll
        for (int r = 0; r < 8; ++r) {
            const int t  = tok0 + r;
            const int ld = t >> 4, lh = (t >> 2) & 3, lw = t & 3;
            const size_t row = ((size_t)((wb * 48 + wd * 4 + ld) * 48 + (wh * 4 + lh)) * 48
                                + (ww * 4 + lw)) * (size_t)C_DIM;
            out[row + col] = acc_out[j][r];
        }
    }
}

extern "C" void kernel_launch(void* const* d_in, const int* in_sizes, int n_in,
                              void* d_out, int out_size, void* d_ws, size_t ws_size,
                              hipStream_t stream) {
    (void)in_sizes; (void)n_in; (void)out_size;
    const float* x    = (const float*)d_in[0];
    const float* lnw  = (const float*)d_in[1];
    const float* lnb  = (const float*)d_in[2];
    const float* wqkv = (const float*)d_in[3];
    const float* rpbt = (const float*)d_in[4];
    const float* wout = (const float*)d_in[5];
    float* out = (float*)d_out;

    // LDS: W(73728) + xn(12288) + q(12288) + k(12288) + vT(12288) + attn(8192) + bias(28)
    const int smem_bytes = 131136;
    const int n_windows  = 2 * GRID_W * GRID_W * GRID_W;   // 3456
    const size_t pack_bytes = (size_t)PACK_TOTAL_HALVES * sizeof(_Float16);  // 221184

    if (ws_size >= pack_bytes) {
        _Float16* wpack = (_Float16*)d_ws;
        swin_pack_weights<<<dim3((PACK_TOTAL_HALVES + 255) / 256), dim3(256), 0, stream>>>(
            wqkv, wout, wpack);
        (void)hipFuncSetAttribute((const void*)swin_win_attn<true>,
                                  hipFuncAttributeMaxDynamicSharedMemorySize, smem_bytes);
        swin_win_attn<true><<<dim3(n_windows), dim3(128), smem_bytes, stream>>>(
            x, lnw, lnb, wqkv, rpbt, wout, wpack, out);
    } else {
        (void)hipFuncSetAttribute((const void*)swin_win_attn<false>,
                                  hipFuncAttributeMaxDynamicSharedMemorySize, smem_bytes);
        swin_win_attn<false><<<dim3(n_windows), dim3(128), smem_bytes, stream>>>(
            x, lnw, lnb, wqkv, rpbt, wout, (const _Float16*)nullptr, out);
    }
}
